// CQAttention_15633680957796
// MI455X (gfx1250) — matrix operand order repaired and checked by reading, not compiled, not run on hardware
//
#include <hip/hip_runtime.h>
#include <cstdint>
#include <cstddef>

// ---------------------------------------------------------------------------
// CQAttention for MI455X (gfx1250).
// GEMMs: v_wmma_f32_16x16x32_bf16. Strided operand tiles staged global->LDS
// with the Tensor Data Mover (tensor_load_to_lds + s_wait_tensorcnt) when the
// toolchain exposes it; cooperative load fallback otherwise.
// B=32, D=128, Lc=2048, Lq=512.
// ---------------------------------------------------------------------------

typedef __attribute__((ext_vector_type(16))) __bf16 v16bf;
typedef __attribute__((ext_vector_type(8)))  float  v8f;
typedef __attribute__((ext_vector_type(4)))  unsigned int u32x4;
typedef __attribute__((ext_vector_type(8)))  int i32x8;
typedef __attribute__((ext_vector_type(4)))  int i32x4;

namespace {
constexpr int kB  = 32;
constexpr int kD  = 128;
constexpr int kLc = 2048;
constexpr int kLq = 512;
constexpr float kNegInf = -1e30f;
}

#if __has_builtin(__builtin_amdgcn_tensor_load_to_lds)
#define CQ_TDM 1
#else
#define CQ_TDM 0
#endif

#if __has_builtin(__builtin_amdgcn_s_wait_tensorcnt)
#define CQ_TDM_WAIT() __builtin_amdgcn_s_wait_tensorcnt(0)
#else
#define CQ_TDM_WAIT() asm volatile("s_wait_tensorcnt 0x0" ::: "memory")
#endif

// bf16 stored as raw u16 in workspace (round-to-nearest-even).
__device__ __forceinline__ unsigned short f2bfr(float f) {
  unsigned u = __builtin_bit_cast(unsigned, f);
  unsigned r = (u + 0x7fffu + ((u >> 16) & 1u)) >> 16;
  return (unsigned short)r;
}
__device__ __forceinline__ __bf16 bfr(unsigned short s) {
  return __builtin_bit_cast(__bf16, s);
}

#if CQ_TDM
// TDM: DMA a 32-row x 16-col bf16 tile (row stride = stride_elems) into LDS.
// D# per CDNA5 ISA ch.8: group0 = {count/type, lds_addr, global_addr},
// group1 = {data_size, tensor dims, tile dims, dim0 stride}. 2-D tensor, so
// descriptor groups 2/3 are zero/disabled.
__device__ __forceinline__ void tdm_tile_32x16(
    const void* lds_dst, const unsigned short* gsrc, unsigned stride_elems) {
  const unsigned long long ga = (unsigned long long)(const void*)gsrc;
  u32x4 g0;
  g0[0] = 1u;                                        // count=1 (valid user D#)
  g0[1] = (unsigned)(unsigned long long)lds_dst;     // lds_addr (bytes)
  g0[2] = (unsigned)(ga & 0xffffffffu);              // global_addr[31:0]
  g0[3] = (unsigned)((ga >> 32) & 0x01ffffffu) | (2u << 30);  // addr[56:32] | type=2
  const unsigned td0 = 1u << 20, td1 = 1u << 20;     // generous dims: no OOB clip
  i32x8 g1;
  g1[0] = (int)(1u << 16);                               // wg_mask=0, data_size=2B
  g1[1] = (int)((td0 & 0xffffu) << 16);                  // tensor_dim0[15:0]
  g1[2] = (int)((td0 >> 16) | ((td1 & 0xffffu) << 16));  // dim0[31:16] | dim1[15:0]
  g1[3] = (int)((td1 >> 16) | (16u << 16));              // dim1[31:16] | tile_dim0=16
  g1[4] = (int)32u;                                      // tile_dim1=32, tile_dim2=0
  g1[5] = (int)stride_elems;                             // tensor_dim0_stride[31:0]
  g1[6] = 0;                                             // stride[47:32] | dim1_stride lo
  g1[7] = 0;
  i32x4 zz = {0, 0, 0, 0};
#if defined(__clang_major__) && (__clang_major__ >= 23)
  i32x8 z8 = {0, 0, 0, 0, 0, 0, 0, 0};
  __builtin_amdgcn_tensor_load_to_lds(g0, g1, zz, zz, z8, 0);
#else
  __builtin_amdgcn_tensor_load_to_lds(g0, g1, zz, zz, 0);
#endif
}
#endif  // CQ_TDM

// ---------------------------------------------------------------------------
// Prep: Ct (bf16), Ct*w4mlu (bf16), sub0 = Ct.w4C    -- coalesced over c
// ---------------------------------------------------------------------------
__global__ __launch_bounds__(256) void cq_prep_c(
    const float* __restrict__ C, const float* __restrict__ w4C,
    const float* __restrict__ w4mlu, unsigned short* __restrict__ cw,
    unsigned short* __restrict__ ctb, float* __restrict__ sub0) {
  const int c = blockIdx.x * 256 + threadIdx.x;
  const int b = blockIdx.y;
  const float* Cb = C + (size_t)b * kD * kLc;
  const size_t ob = ((size_t)b * kLc + c) * kD;
  float s0 = 0.f;
  #pragma unroll 4
  for (int d = 0; d < kD; ++d) {
    const float v = Cb[(size_t)d * kLc + c];
    s0 += v * w4C[d];
    cw[ob + d]  = f2bfr(v * w4mlu[d]);
    ctb[ob + d] = f2bfr(v);
  }
  sub0[b * kLc + c] = s0;
}

__global__ __launch_bounds__(256) void cq_prep_q(
    const float* __restrict__ Q, const float* __restrict__ w4Q,
    unsigned short* __restrict__ qtb, float* __restrict__ sub1) {
  const int q = blockIdx.x * 256 + threadIdx.x;
  const int b = blockIdx.y;
  const float* Qb = Q + (size_t)b * kD * kLq;
  const size_t ob = ((size_t)b * kLq + q) * kD;
  float s = 0.f;
  #pragma unroll 4
  for (int d = 0; d < kD; ++d) {
    const float v = Qb[(size_t)d * kLq + q];
    s += v * w4Q[d];
    qtb[ob + d] = f2bfr(v);
  }
  sub1[b * kLq + q] = s;
}

// ---------------------------------------------------------------------------
// GEMM 1: S[b,c,q] = sum_d (Ct*w4mlu)[c,d] * Qt[q,d] + sub0[c] + sub1[q] + bias
// Both operands row-major in d => contiguous fragment loads (b128-friendly).
// ---------------------------------------------------------------------------
__global__ __launch_bounds__(128) void cq_gemm_s(
    const unsigned short* __restrict__ cw, const unsigned short* __restrict__ qt,
    const float* __restrict__ sub0, const float* __restrict__ sub1,
    const float* __restrict__ bias, float* __restrict__ S) {
  const int lane = threadIdx.x & 31, wv = threadIdx.x >> 5;
  const int l16 = lane & 15, half = lane >> 4;
  const int b  = blockIdx.z;
  const int q0 = blockIdx.x * 16;
  const int c0 = (blockIdx.y * 4 + wv) * 16;
  const unsigned short* Ap = cw + ((size_t)b * kLc + c0 + l16) * kD;
  const unsigned short* Bp = qt + ((size_t)b * kLq + q0 + l16) * kD + (half << 4);
  v8f acc = {};
  #pragma unroll
  for (int kk = 0; kk < kD; kk += 32) {
    v16bf a, bm;
    #pragma unroll
    for (int e = 0; e < 16; ++e) {
      const int p = e >> 1, bit = e & 1;
      const int ka = e + (half << 4); // A 16x32
      a[e]  = bfr(Ap[kk + ka]);
      bm[e] = bfr(Bp[kk + e]);                                            // B 32x16
    }
    acc = __builtin_amdgcn_wmma_f32_16x16x32_bf16(false, a, false, bm, (short)0,
                                                  acc, false, false);
  }
  const float add_q = sub1[b * kLq + q0 + l16] + bias[0];
  #pragma unroll
  for (int v = 0; v < 8; ++v) {
    const int c = c0 + v + (half << 3);
    S[((size_t)b * kLc + c) * kLq + q0 + l16] = acc[v] + sub0[b * kLc + c] + add_q;
  }
}

// ---------------------------------------------------------------------------
// Softmax statistics (row = axis q with Qmask, col = axis c with Cmask)
// ---------------------------------------------------------------------------
__global__ __launch_bounds__(256) void cq_rowstats(
    const float* __restrict__ S, const float* __restrict__ Qmask,
    float* __restrict__ rmax, float* __restrict__ rsum) {
  const int row = blockIdx.x;                 // b*kLc + c
  const int b = row / kLc;
  const float* Sr = S + (size_t)row * kLq;
  const float* Qm = Qmask + b * kLq;
  const int t = threadIdx.x;
  float m = kNegInf;
  for (int q = t; q < kLq; q += 256) {
    const float mk = Qm[q];
    m = fmaxf(m, Sr[q] * mk + (1.f - mk) * kNegInf);
  }
  __shared__ float red[256];
  red[t] = m; __syncthreads();
  for (int s2 = 128; s2 > 0; s2 >>= 1) { if (t < s2) red[t] = fmaxf(red[t], red[t + s2]); __syncthreads(); }
  const float gm = red[0]; __syncthreads();
  float sum = 0.f;
  for (int q = t; q < kLq; q += 256) {
    const float mk = Qm[q];
    sum += __expf(Sr[q] * mk + (1.f - mk) * kNegInf - gm);
  }
  red[t] = sum; __syncthreads();
  for (int s2 = 128; s2 > 0; s2 >>= 1) { if (t < s2) red[t] += red[t + s2]; __syncthreads(); }
  if (t == 0) { rmax[row] = gm; rsum[row] = red[0]; }
}

__global__ __launch_bounds__(256) void cq_colstats(
    const float* __restrict__ S, const float* __restrict__ Cmask,
    float* __restrict__ cmax, float* __restrict__ csum) {
  const int col = blockIdx.x;                 // b*kLq + q
  const int b = col / kLq, q = col % kLq;
  const float* Sb = S + (size_t)b * kLc * kLq + q;
  const float* Cm = Cmask + b * kLc;
  const int t = threadIdx.x;
  float m = kNegInf;
  for (int c = t; c < kLc; c += 256) {
    const float mk = Cm[c];
    m = fmaxf(m, Sb[(size_t)c * kLq] * mk + (1.f - mk) * kNegInf);
  }
  __shared__ float red[256];
  red[t] = m; __syncthreads();
  for (int s2 = 128; s2 > 0; s2 >>= 1) { if (t < s2) red[t] = fmaxf(red[t], red[t + s2]); __syncthreads(); }
  const float gm = red[0]; __syncthreads();
  float sum = 0.f;
  for (int c = t; c < kLc; c += 256) {
    const float mk = Cm[c];
    sum += __expf(Sb[(size_t)c * kLq] * mk + (1.f - mk) * kNegInf - gm);
  }
  red[t] = sum; __syncthreads();
  for (int s2 = 128; s2 > 0; s2 >>= 1) { if (t < s2) red[t] += red[t + s2]; __syncthreads(); }
  if (t == 0) { cmax[col] = gm; csum[col] = red[0]; }
}

__global__ __launch_bounds__(256) void cq_softmax_norm(
    const float* __restrict__ S, const float* __restrict__ Cmask,
    const float* __restrict__ Qmask,
    const float* __restrict__ rmax, const float* __restrict__ rsum,
    const float* __restrict__ cmax, const float* __restrict__ csum,
    unsigned short* __restrict__ S1, unsigned short* __restrict__ S2) {
  const size_t i = (size_t)blockIdx.x * 256 + threadIdx.x;
  const int q = (int)(i % kLq);
  const size_t bc = i / kLq;                  // b*kLc + c
  const int c = (int)(bc % kLc);
  const int b = (int)(bc / kLc);
  const float s  = S[i];
  const float qm = Qmask[b * kLq + q];
  const float cm = Cmask[b * kLc + c];
  const float e1 = __expf(s * qm + (1.f - qm) * kNegInf - rmax[bc]) / rsum[bc];
  const float e2 = __expf(s * cm + (1.f - cm) * kNegInf - cmax[b * kLq + q]) / csum[b * kLq + q];
  S1[i] = f2bfr(e1);
  S2[i] = f2bfr(e2);
}

// ---------------------------------------------------------------------------
// GEMM 2: T[b,q,d] = sum_c S2[b,c,q] * Ct[b,c,d]   (K = Lc = 2048)
// Both operands strided in K => stage 32x16 tiles in LDS via TDM.
// Waves share the A (S2) tile; each wave owns a 16-wide d-slice (B tile).
// ---------------------------------------------------------------------------
__global__ __launch_bounds__(128) void cq_gemm_t(
    const unsigned short* __restrict__ S2, const unsigned short* __restrict__ ctb,
    unsigned short* __restrict__ T) {
  __shared__ unsigned short ldsA[512];        // S2 tile: 32 c-rows x 16 q-cols
  __shared__ unsigned short ldsB[4][512];     // per-wave Ct tile: 32 c x 16 d
  const int lane = threadIdx.x & 31, wv = threadIdx.x >> 5;
  const int l16 = lane & 15, half = lane >> 4;
  const int b  = blockIdx.z;
  const int q0 = blockIdx.x * 16;
  const int d0 = (blockIdx.y * 4 + wv) * 16;
  const unsigned short* Ag = S2 + (size_t)b * kLc * kLq + q0;   // row c: + c*kLq
  const unsigned short* Bg = ctb + (size_t)b * kLc * kD + d0;   // row c: + c*kD
  v8f acc = {};
  for (int kk = 0; kk < kLc; kk += 32) {
#if CQ_TDM
    if (wv == 0) tdm_tile_32x16(ldsA, Ag + (size_t)kk * kLq, kLq);
    tdm_tile_32x16(&ldsB[wv][0], Bg + (size_t)kk * kD, kD);
    CQ_TDM_WAIT();
#else
    for (int j = threadIdx.x; j < 512; j += 128)
      ldsA[j] = Ag[(size_t)(kk + (j >> 4)) * kLq + (j & 15)];
    for (int w = 0; w < 4; ++w) {
      const unsigned short* Bgw =
          ctb + (size_t)b * kLc * kD + (blockIdx.y * 4 + w) * 16;
      for (int j = threadIdx.x; j < 512; j += 128)
        ldsB[w][j] = Bgw[(size_t)(kk + (j >> 4)) * kD + (j & 15)];
    }
#endif
    __syncthreads();
    v16bf a, bm;
    #pragma unroll
    for (int e = 0; e < 16; ++e) {
      const int p = e >> 1, bit = e & 1;
      const int ka = e + (half << 4);
      const int kb = e + (half << 4);
      a[e]  = bfr(ldsA[ka * 16 + l16]);        // A[m=q, k=c]
      bm[e] = bfr(ldsB[wv][kb * 16 + l16]);    // B[k=c, n=d]
    }
    acc = __builtin_amdgcn_wmma_f32_16x16x32_bf16(false, a, false, bm, (short)0,
                                                  acc, false, false);
    __syncthreads();
  }
  #pragma unroll
  for (int v = 0; v < 8; ++v) {
    const int q = q0 + v + (half << 3);
    T[((size_t)b * kLq + q) * kD + d0 + l16] = f2bfr(acc[v]);
  }
}

// ---------------------------------------------------------------------------
// GEMM 3+4 fused: A = S1*Qt, Bmat = S1*T share the A-fragment (S1, contiguous
// global loads). The two shared B tiles (Qt, T) are TDM-staged by wave 0.
// Epilogue writes out channels [D,2D)=A, [2D,3D)=Ct*A, [3D,4D)=Ct*Bmat.
// ---------------------------------------------------------------------------
__global__ __launch_bounds__(128) void cq_gemm_ab(
    const unsigned short* __restrict__ S1, const unsigned short* __restrict__ qt,
    const unsigned short* __restrict__ T, const float* __restrict__ C,
    float* __restrict__ out) {
  __shared__ unsigned short ldsQ[512];        // Qt tile: 32 q-rows x 16 d-cols
  __shared__ unsigned short ldsT[512];        // T  tile: 32 q-rows x 16 d-cols
  const int lane = threadIdx.x & 31, wv = threadIdx.x >> 5;
  const int l16 = lane & 15, half = lane >> 4;
  const int b  = blockIdx.z;
  const int d0 = blockIdx.x * 16;
  const int c0 = (blockIdx.y * 4 + wv) * 16;
  const unsigned short* Ap = S1 + ((size_t)b * kLc + c0 + l16) * kLq;
  const unsigned short* Qg = qt + (size_t)b * kLq * kD + d0;    // row q: + q*kD
  const unsigned short* Tg = T  + (size_t)b * kLq * kD + d0;
  v8f accA = {}, accB = {};
  for (int kk = 0; kk < kLq; kk += 32) {
#if CQ_TDM
    if (wv == 0) {
      tdm_tile_32x16(ldsQ, Qg + (size_t)kk * kD, kD);
      tdm_tile_32x16(ldsT, Tg + (size_t)kk * kD, kD);
      CQ_TDM_WAIT();
    }
#else
    for (int j = threadIdx.x; j < 512; j += 128) {
      const size_t r = (size_t)(kk + (j >> 4)) * kD + (j & 15);
      ldsQ[j] = Qg[r];
      ldsT[j] = Tg[r];
    }
#endif
    __syncthreads();
    v16bf a, bq, bt;
    #pragma unroll
    for (int e = 0; e < 16; ++e) {
      const int p = e >> 1, bit = e & 1;
      const int ka = e + (half << 4);
      const int kb = e + (half << 4);
      a[e]  = bfr(Ap[kk + ka]);                // A[m=c, k=q], contiguous pairs
      bq[e] = bfr(ldsQ[kb * 16 + l16]);        // B[k=q, n=d]
      bt[e] = bfr(ldsT[kb * 16 + l16]);
    }
    accA = __builtin_amdgcn_wmma_f32_16x16x32_bf16(false, a, false, bq, (short)0,
                                                   accA, false, false);
    accB = __builtin_amdgcn_wmma_f32_16x16x32_bf16(false, a, false, bt, (short)0,
                                                   accB, false, false);
    __syncthreads();
  }
  const int d = d0 + l16;
  const size_t ob = (size_t)b * 4 * kD * kLc;
  #pragma unroll
  for (int v = 0; v < 8; ++v) {
    const int c = c0 + v + (half << 3);
    const float ctv = C[((size_t)b * kD + d) * kLc + c];
    const float av = accA[v], bv = accB[v];
    out[ob + (size_t)(kD + d) * kLc + c]     = av;
    out[ob + (size_t)(2 * kD + d) * kLc + c] = ctv * av;
    out[ob + (size_t)(3 * kD + d) * kLc + c] = ctv * bv;
  }
}

// Channels [0,D) of output are exactly C.
__global__ __launch_bounds__(256) void cq_copy_ct(
    const float* __restrict__ C, float* __restrict__ out) {
  const size_t i = (size_t)blockIdx.x * 256 + threadIdx.x;   // over B*D*Lc
  const int c = (int)(i % kLc);
  const size_t bd = i / kLc;
  const int d = (int)(bd % kD);
  const int b = (int)(bd / kD);
  out[((size_t)b * 4 * kD + d) * kLc + c] = C[i];
}

// ---------------------------------------------------------------------------
extern "C" void kernel_launch(void* const* d_in, const int* in_sizes, int n_in,
                              void* d_out, int out_size, void* d_ws, size_t ws_size,
                              hipStream_t stream) {
  (void)in_sizes; (void)n_in; (void)out_size; (void)ws_size;
  const float* C     = (const float*)d_in[0];
  const float* Q     = (const float*)d_in[1];
  const float* Cmask = (const float*)d_in[2];
  const float* Qmask = (const float*)d_in[3];
  const float* w4C   = (const float*)d_in[4];
  const float* w4Q   = (const float*)d_in[5];
  const float* w4mlu = (const float*)d_in[6];
  const float* bias  = (const float*)d_in[7];
  float* out = (float*)d_out;

  char* ws = (char*)d_ws;
  size_t off = 0;
  auto carve = [&](size_t bytes) -> char* {
    char* p = ws + off;
    off += (bytes + 255) & ~(size_t)255;
    return p;
  };
  float*          Sbuf = (float*)         carve((size_t)kB * kLc * kLq * 4); // 134 MB
  unsigned short* S1   = (unsigned short*)carve((size_t)kB * kLc * kLq * 2); //  67 MB
  unsigned short* S2   = (unsigned short*)carve((size_t)kB * kLc * kLq * 2); //  67 MB
  unsigned short* cw   = (unsigned short*)carve((size_t)kB * kLc * kD * 2);
  unsigned short* ctb  = (unsigned short*)carve((size_t)kB * kLc * kD * 2);
  unsigned short* qtb  = (unsigned short*)carve((size_t)kB * kLq * kD * 2);
  unsigned short* Tb   = (unsigned short*)carve((size_t)kB * kLq * kD * 2);
  float* sub0 = (float*)carve((size_t)kB * kLc * 4);
  float* sub1 = (float*)carve((size_t)kB * kLq * 4);
  float* rmax = (float*)carve((size_t)kB * kLc * 4);
  float* rsum = (float*)carve((size_t)kB * kLc * 4);
  float* cmax = (float*)carve((size_t)kB * kLq * 4);
  float* csum = (float*)carve((size_t)kB * kLq * 4);

  cq_prep_c<<<dim3(kLc / 256, kB), 256, 0, stream>>>(C, w4C, w4mlu, cw, ctb, sub0);
  cq_prep_q<<<dim3(kLq / 256, kB), 256, 0, stream>>>(Q, w4Q, qtb, sub1);
  cq_gemm_s<<<dim3(kLq / 16, kLc / 64, kB), 128, 0, stream>>>(cw, qtb, sub0, sub1, bias, Sbuf);
  cq_rowstats<<<kB * kLc, 256, 0, stream>>>(Sbuf, Qmask, rmax, rsum);
  cq_colstats<<<kB * kLq, 256, 0, stream>>>(Sbuf, Cmask, cmax, csum);
  cq_softmax_norm<<<(unsigned)(((size_t)kB * kLc * kLq) / 256), 256, 0, stream>>>(
      Sbuf, Cmask, Qmask, rmax, rsum, cmax, csum, S1, S2);
  cq_gemm_t<<<dim3(kLq / 16, kD / 64, kB), 128, 0, stream>>>(S2, ctb, Tb);
  cq_gemm_ab<<<dim3(kD / 16, kLc / 64, kB), 128, 0, stream>>>(S1, qtb, Tb, C, out);
  cq_copy_ct<<<(unsigned)(((size_t)kB * kD * kLc) / 256), 256, 0, stream>>>(C, out);
}
